// DictionaryLearningLayer_1589137900197
// MI455X (gfx1250) — matrix operand order repaired and checked by reading, not compiled, not run on hardware
//
#include <hip/hip_runtime.h>

// Problem constants (match reference).
#define B_SZ 16384
#define M_SZ 512
#define N_SZ 2048
#define N_ITERS 10

typedef __attribute__((ext_vector_type(16))) __bf16 v16bf;
typedef __attribute__((ext_vector_type(8)))  __bf16 v8bf;
typedef __attribute__((ext_vector_type(4)))  __bf16 v4bf;
typedef __attribute__((ext_vector_type(8)))  float  v8f;
typedef __attribute__((ext_vector_type(4)))  float  v4f;

// ---------------- helpers ----------------

// Load two contiguous 8xbf16 chunks (16B each) into one 16-element fragment.
__device__ __forceinline__ v16bf ld2x8(const __bf16* p0, const __bf16* p1) {
    v8bf lo = *(const v8bf*)p0;
    v8bf hi = *(const v8bf*)p1;
    return __builtin_shufflevector(lo, hi, 0,1,2,3,4,5,6,7,8,9,10,11,12,13,14,15);
}

// Load two 8xf32 chunks and convert to a bf16 fragment (A operand of GEMM2).
__device__ __forceinline__ v16bf ldcvt2x8(const float* p0, const float* p1) {
    v4f f0 = *(const v4f*)p0;
    v4f f1 = *(const v4f*)(p0 + 4);
    v4f f2 = *(const v4f*)p1;
    v4f f3 = *(const v4f*)(p1 + 4);
    v4bf b0 = __builtin_convertvector(f0, v4bf);
    v4bf b1 = __builtin_convertvector(f1, v4bf);
    v4bf b2 = __builtin_convertvector(f2, v4bf);
    v4bf b3 = __builtin_convertvector(f3, v4bf);
    v8bf lo = __builtin_shufflevector(b0, b1, 0,1,2,3,4,5,6,7);
    v8bf hi = __builtin_shufflevector(b2, b3, 0,1,2,3,4,5,6,7);
    return __builtin_shufflevector(lo, hi, 0,1,2,3,4,5,6,7,8,9,10,11,12,13,14,15);
}

#define WMMA_BF16(A, Bf, C) \
    __builtin_amdgcn_wmma_f32_16x16x32_bf16(false, (A), false, (Bf), (short)0, (C), false, false)

// ---------------- init kernels ----------------

__global__ void init_v_kernel(const float* __restrict__ batch,
                              float* __restrict__ v, __bf16* __restrict__ vb) {
    size_t i = (size_t)blockIdx.x * blockDim.x + threadIdx.x;
    float f = batch[i];
    v[i]  = f;
    vb[i] = (__bf16)f;
}

__global__ void init_d_kernel(const float* __restrict__ D,
                              __bf16* __restrict__ Db, __bf16* __restrict__ Dtb) {
    size_t i = (size_t)blockIdx.x * blockDim.x + threadIdx.x;   // i = m*N + n
    int m = (int)(i / N_SZ);
    int n = (int)(i % N_SZ);
    float f = D[i];
    Db[i] = (__bf16)f;
    Dtb[(size_t)n * M_SZ + m] = (__bf16)f;
}

// ---------------- row-norm kernels (one wave per row) ----------------

// t[b] = gamma/sqrt(M) * ||v[b,:]||_2
__global__ void rownorm_v_kernel(const float* __restrict__ v, float* __restrict__ tbuf,
                                 const float* __restrict__ gamma_p) {
    int wave = threadIdx.x >> 5;
    int lane = threadIdx.x & 31;
    int row  = blockIdx.x * 8 + wave;
    const float* p = v + (size_t)row * M_SZ;
    float s = 0.f;
    #pragma unroll
    for (int i = lane; i < M_SZ; i += 32) { float x = p[i]; s += x * x; }
    #pragma unroll
    for (int m = 16; m >= 1; m >>= 1) s += __shfl_xor(s, m, 32);
    if (lane == 0)
        tbuf[row] = (*gamma_p) * __builtin_sqrtf(s * (1.0f / (float)M_SZ));
}

// s[b] = (beta/M) * ||alpha[b,:]||_2
__global__ void rownorm_a_kernel(const float* __restrict__ alpha, float* __restrict__ sbuf,
                                 const float* __restrict__ beta_p) {
    int wave = threadIdx.x >> 5;
    int lane = threadIdx.x & 31;
    int row  = blockIdx.x * 8 + wave;
    const float* p = alpha + (size_t)row * N_SZ;
    float s = 0.f;
    #pragma unroll 4
    for (int i = lane; i < N_SZ; i += 32) { float x = p[i]; s += x * x; }
    #pragma unroll
    for (int m = 16; m >= 1; m >>= 1) s += __shfl_xor(s, m, 32);
    if (lane == 0)
        sbuf[row] = (*beta_p) * (1.0f / (float)M_SZ) * __builtin_sqrtf(s);
}

// ---------------- GEMM1: alpha_new = relu(alpha + beta*(v@D) - t[row]) ----------------
// A = vb [B, M] bf16 (K = M = 512), B = Dt_bf16 [N, M] (column n contiguous over K).
// Wave computes 16 rows x 64 cols. Block = 8 waves -> 128 rows x 64 cols.

__global__ __launch_bounds__(256)
void amp_gemm1_kernel(const __bf16* __restrict__ vb, const __bf16* __restrict__ Dtb,
                      float* __restrict__ alpha, const float* __restrict__ tbuf,
                      const float* __restrict__ beta_p, int first) {
    const int lane = threadIdx.x & 31;
    const int wave = threadIdx.x >> 5;
    const int hi   = lane >> 4;        // half-wave select
    const int l15  = lane & 15;
    const int rowBase = blockIdx.y * 128 + wave * 16;
    const int colBase = blockIdx.x * 64;
    const float beta = *beta_p;

    const int aoff = hi ? 8  : 0;      // C: lanes16-31 hold rows M+8..M+15
    const int boff = hi ? 16 : 0;      // B: lanes16-31 hold K+16..K+31

    const __bf16* ap  = vb  + (size_t)(rowBase + l15) * M_SZ;
    const __bf16* bp0 = Dtb + (size_t)(colBase +  0 + l15) * M_SZ;
    const __bf16* bp1 = Dtb + (size_t)(colBase + 16 + l15) * M_SZ;
    const __bf16* bp2 = Dtb + (size_t)(colBase + 32 + l15) * M_SZ;
    const __bf16* bp3 = Dtb + (size_t)(colBase + 48 + l15) * M_SZ;

    v8f c0 = {}, c1 = {}, c2 = {}, c3 = {};

    #pragma unroll 4
    for (int k = 0; k < M_SZ; k += 32) {
        v16bf a  = ld2x8(ap + k + boff, ap + k + boff + 8);
        v16bf b0 = ld2x8(bp0 + k + boff, bp0 + k + boff + 8);
        v16bf b1 = ld2x8(bp1 + k + boff, bp1 + k + boff + 8);
        v16bf b2 = ld2x8(bp2 + k + boff, bp2 + k + boff + 8);
        v16bf b3 = ld2x8(bp3 + k + boff, bp3 + k + boff + 8);
        c0 = WMMA_BF16(a, b0, c0);
        c1 = WMMA_BF16(a, b1, c1);
        c2 = WMMA_BF16(a, b2, c2);
        c3 = WMMA_BF16(a, b3, c3);
    }

    // Epilogue: z = alpha_old + beta*acc ; alpha_new = max(z - t, 0)
    const int col0 = colBase + l15;
    #pragma unroll
    for (int g = 0; g < 8; ++g) {
        int row = rowBase + g + aoff;                // C layout: M = g + 8*hi
        float tt = tbuf[row];
        size_t base = (size_t)row * N_SZ + col0;
        float z0 = (first ? 0.f : alpha[base +  0]) + beta * c0[g];
        float z1 = (first ? 0.f : alpha[base + 16]) + beta * c1[g];
        float z2 = (first ? 0.f : alpha[base + 32]) + beta * c2[g];
        float z3 = (first ? 0.f : alpha[base + 48]) + beta * c3[g];
        alpha[base +  0] = fmaxf(z0 - tt, 0.f);
        alpha[base + 16] = fmaxf(z1 - tt, 0.f);
        alpha[base + 32] = fmaxf(z2 - tt, 0.f);
        alpha[base + 48] = fmaxf(z3 - tt, 0.f);
    }
}

// ---------------- GEMM2: v_new = y - beta*(alpha @ D^T) + s[row]*v_old ----------------
// A = alpha [B, N] f32 converted on the fly (K = N = 2048), B = D_bf16 [M, N]
// (column m of D^T contiguous over K as row m of D). Wave: 16 rows x 64 cols of M.

__global__ __launch_bounds__(256)
void amp_gemm2_kernel(const float* __restrict__ alpha, const __bf16* __restrict__ Db,
                      const float* __restrict__ y, float* __restrict__ v,
                      __bf16* __restrict__ vb, const float* __restrict__ sbuf,
                      const float* __restrict__ beta_p) {
    const int lane = threadIdx.x & 31;
    const int wave = threadIdx.x >> 5;
    const int hi   = lane >> 4;
    const int l15  = lane & 15;
    const int rowBase = blockIdx.y * 128 + wave * 16;
    const int colBase = blockIdx.x * 64;
    const float beta = *beta_p;

    const int aoff = hi ? 8  : 0;
    const int boff = hi ? 16 : 0;

    const float*  ap  = alpha + (size_t)(rowBase + l15) * N_SZ;
    const __bf16* bp0 = Db + (size_t)(colBase +  0 + l15) * N_SZ;
    const __bf16* bp1 = Db + (size_t)(colBase + 16 + l15) * N_SZ;
    const __bf16* bp2 = Db + (size_t)(colBase + 32 + l15) * N_SZ;
    const __bf16* bp3 = Db + (size_t)(colBase + 48 + l15) * N_SZ;

    v8f c0 = {}, c1 = {}, c2 = {}, c3 = {};

    #pragma unroll 2
    for (int k = 0; k < N_SZ; k += 32) {
        v16bf a  = ldcvt2x8(ap + k + boff, ap + k + boff + 8);
        v16bf b0 = ld2x8(bp0 + k + boff, bp0 + k + boff + 8);
        v16bf b1 = ld2x8(bp1 + k + boff, bp1 + k + boff + 8);
        v16bf b2 = ld2x8(bp2 + k + boff, bp2 + k + boff + 8);
        v16bf b3 = ld2x8(bp3 + k + boff, bp3 + k + boff + 8);
        c0 = WMMA_BF16(a, b0, c0);
        c1 = WMMA_BF16(a, b1, c1);
        c2 = WMMA_BF16(a, b2, c2);
        c3 = WMMA_BF16(a, b3, c3);
    }

    const int col0 = colBase + l15;
    #pragma unroll
    for (int g = 0; g < 8; ++g) {
        int row = rowBase + g + aoff;
        float sc = sbuf[row];
        size_t base = (size_t)row * M_SZ + col0;
        float n0 = y[base +  0] - beta * c0[g] + sc * v[base +  0];
        float n1 = y[base + 16] - beta * c1[g] + sc * v[base + 16];
        float n2 = y[base + 32] - beta * c2[g] + sc * v[base + 32];
        float n3 = y[base + 48] - beta * c3[g] + sc * v[base + 48];
        v[base +  0] = n0;  vb[base +  0] = (__bf16)n0;
        v[base + 16] = n1;  vb[base + 16] = (__bf16)n1;
        v[base + 32] = n2;  vb[base + 32] = (__bf16)n2;
        v[base + 48] = n3;  vb[base + 48] = (__bf16)n3;
    }
}

// ---------------- host-side orchestration ----------------

extern "C" void kernel_launch(void* const* d_in, const int* in_sizes, int n_in,
                              void* d_out, int out_size, void* d_ws, size_t ws_size,
                              hipStream_t stream) {
    const float* batch   = (const float*)d_in[0];   // [B, M]
    const float* D       = (const float*)d_in[1];   // [M, N]
    const float* gamma_p = (const float*)d_in[2];
    const float* beta_p  = (const float*)d_in[3];
    float* alpha = (float*)d_out;                   // [B, N] — working + final output

    // Workspace carve-out (all 256B aligned): ~54 MB total.
    char* ws = (char*)d_ws;
    size_t off = 0;
    auto carve = [&](size_t bytes) -> void* {
        void* p = ws + off;
        off = (off + bytes + 255) & ~(size_t)255;
        return p;
    };
    float*  vbuf = (float*) carve((size_t)B_SZ * M_SZ * 4);   // v master (f32)
    __bf16* vb   = (__bf16*)carve((size_t)B_SZ * M_SZ * 2);   // v shadow (bf16)
    __bf16* Db   = (__bf16*)carve((size_t)M_SZ * N_SZ * 2);   // D bf16
    __bf16* Dtb  = (__bf16*)carve((size_t)M_SZ * N_SZ * 2);   // D^T bf16
    float*  tbuf = (float*) carve((size_t)B_SZ * 4);          // per-row threshold
    float*  sbuf = (float*) carve((size_t)B_SZ * 4);          // per-row ||alpha|| coeff

    init_v_kernel<<<(B_SZ * M_SZ) / 256, 256, 0, stream>>>(batch, vbuf, vb);
    init_d_kernel<<<(M_SZ * N_SZ) / 256, 256, 0, stream>>>(D, Db, Dtb);

    dim3 g1(N_SZ / 64, B_SZ / 128);   // 32 x 128
    dim3 g2(M_SZ / 64, B_SZ / 128);   //  8 x 128

    for (int it = 0; it < N_ITERS; ++it) {
        rownorm_v_kernel<<<B_SZ / 8, 256, 0, stream>>>(vbuf, tbuf, gamma_p);
        amp_gemm1_kernel<<<g1, 256, 0, stream>>>(vb, Dtb, alpha, tbuf, beta_p, it == 0 ? 1 : 0);
        if (it < N_ITERS - 1) {   // final iteration's v-update is dead code in the reference
            rownorm_a_kernel<<<B_SZ / 8, 256, 0, stream>>>(alpha, sbuf, beta_p);
            amp_gemm2_kernel<<<g2, 256, 0, stream>>>(alpha, Db, batch, vbuf, vb, sbuf, beta_p);
        }
    }
}
